// GraphAttentionLayer2_60662118088930
// MI455X (gfx1250) — matrix operand order repaired and checked by reading, not compiled, not run on hardware
//
#include <hip/hip_runtime.h>

// ---------------------------------------------------------------------------
// GAT layer for MI455X (gfx1250, wave32, WMMA).
// B=8, N=2048, Fin=256, Fout=128.
// Both GEMMs use v_wmma_f32_16x16x32_bf16 (bf16 in, f32 accumulate).
// adj (128 MB) is the bandwidth term: read once per row-tile, scores held in
// LDS (320 KB/WGP: 128 KB f32 e-buffer + 65 KB bf16 prob buffer).
// ---------------------------------------------------------------------------

typedef __attribute__((ext_vector_type(16))) __bf16 v16bf;
typedef __attribute__((ext_vector_type(2)))  __bf16 v2bf;
typedef __attribute__((ext_vector_type(8)))  float  v8f;

#define B_    8
#define N_    2048
#define FIN   256
#define FOUT  128
#define NEGINF (-1000000000.0f)
#define ALPHA  0.2f

// bf16 prob-row stride: 2088 ushorts = 1044 dwords per row.
// 1044 mod 64 = 20 = 4*5, gcd(5,16)=1  ->  the 16 lanes' 4-dword ds_load_b128
// spans occupy 64 distinct LDS banks (conflict-free), and every fragment load
// is 16B aligned (lm*1044 and kt/2 + 8*hlf (+4) are multiples of 4 dwords).
#define PSTRIDE_U16 2088
#define PSTRIDE_U32 1044

__device__ __forceinline__ float bf2f(unsigned short s) {
  unsigned u = ((unsigned)s) << 16;
  return __builtin_bit_cast(float, u);
}

// pack two f32 -> one dword of 2 bf16 (lo = first arg).
// Fallback: round-half-up (u + 0x8000) then pack both high halves with a
// single v_perm_b32 (result bytes = {hi.b3, hi.b2, lo.b3, lo.b2}).
// 3 VALU per pair vs ~6 for the exact-RNE sequence; differs from RNE only on
// exact ties, well below bf16 quantization error.
__device__ __forceinline__ unsigned pack_bf16(float lo, float hi) {
#if __has_builtin(__builtin_amdgcn_cvt_pk_bf16_f32)
  v2bf v = __builtin_amdgcn_cvt_pk_bf16_f32(lo, hi);   // single v_cvt_pk_bf16_f32
  return __builtin_bit_cast(unsigned, v);
#else
  unsigned ul = __builtin_bit_cast(unsigned, lo) + 0x8000u;
  unsigned uh = __builtin_bit_cast(unsigned, hi) + 0x8000u;
  return __builtin_amdgcn_perm(uh, ul, 0x07060302u);   // single v_perm_b32
#endif
}

union AFrag { unsigned u[8]; v16bf v; };      // 16 bf16 elements
union Pack8 { unsigned u[4]; uint4 q; };

// ---------------------------------------------------------------------------
// Kernel 0: W (Fin x Fout, f32) -> WT (Fout x Fin, bf16), pair-packed stores
// ---------------------------------------------------------------------------
__global__ void k_wt(const float* __restrict__ W, unsigned* __restrict__ WT32) {
  int idx = blockIdx.x * 256 + threadIdx.x;   // 0 .. 16383 (uint slots)
  int f  = idx >> 7;                           // Fout index
  int kp = idx & 127;                          // Fin pair index
  WT32[idx] = pack_bf16(W[(2 * kp) * FOUT + f], W[(2 * kp + 1) * FOUT + f]);
}

// ---------------------------------------------------------------------------
// Kernel 1: WhT[b][f][n] = (h @ W)^T  in bf16, via WMMA bf16.
// Grid: 128 blocks x 256 threads; wave w handles a 16-row tile, computing all
// 8 column tiles (Fout=128) so the A fragment is reused 8x.
// ---------------------------------------------------------------------------
__global__ void k_gemm1(const float* __restrict__ h,
                        const unsigned short* __restrict__ WT,
                        unsigned short* __restrict__ WhT) {
  const int t    = threadIdx.x;
  const int wave = t >> 5;
  const int lane = t & 31;
  const int hlf  = lane >> 4;                  // lane half (0/1)
  const int lm   = lane & 15;

  const int b    = blockIdx.x >> 4;            // 16 blocks per batch
  const int row0 = ((blockIdx.x & 15) * 8 + wave) << 4;
  const int row  = row0 + lm;                  // A-matrix M = lane%16

  const float* hrow = h + ((size_t)(b * N_ + row)) * FIN;

  v8f acc[8] = {};

  for (int kt = 0; kt < FIN; kt += 32) {
    // A fragment: 16x32 bf16.  element j -> K = 16*hlf + j
    AFrag a;
    const float2* hp0 = (const float2*)(hrow + kt + 16 * hlf);
    const float2* hp1 = (const float2*)(hrow + kt + 16 * hlf + 8);
#pragma unroll
    for (int q = 0; q < 4; ++q) {
      float2 x = hp0[q], y = hp1[q];
      a.u[q]     = pack_bf16(x.x, x.y);
      a.u[q + 4] = pack_bf16(y.x, y.y);
    }
#pragma unroll
    for (int c = 0; c < 8; ++c) {
      // B fragment: 32x16 bf16. lane = N, element j -> K = 16*hlf + j.
      // WT is (Fout x Fin) so each lane loads 16 contiguous bf16 (32B).
      const v16bf bf = *(const v16bf*)(WT + (size_t)(c * 16 + lm) * FIN + kt + 16 * hlf);
      acc[c] = __builtin_amdgcn_wmma_f32_16x16x32_bf16(
          false, a.v, false, bf, (short)0, acc[c], false, false);
    }
  }

  // C layout: vgpr r, lane -> M = r + 8*hlf, N = lane%16.
  // Store transposed: WhT[b][f = 16c + N][n = row0 + M]; r-varying -> 8
  // consecutive bf16 = one 16B store.
#pragma unroll
  for (int c = 0; c < 8; ++c) {
    Pack8 p;
#pragma unroll
    for (int r = 0; r < 4; ++r) p.u[r] = pack_bf16(acc[c][2 * r], acc[c][2 * r + 1]);
    const int f = c * 16 + lm;
    const int n = row0 + 8 * hlf;
    *(uint4*)(WhT + (((size_t)(b * FOUT + f)) << 11) + n) = p.q;
  }
}

// ---------------------------------------------------------------------------
// Kernel 2: e1[b,n] = Wh[b,n,:].a1 ; e2[b,n] = Wh[b,n,:].a2  (coalesced over n)
// ---------------------------------------------------------------------------
__global__ void k_e(const unsigned short* __restrict__ WhT,
                    const float* __restrict__ a,
                    float* __restrict__ e1, float* __restrict__ e2) {
  int idx = blockIdx.x * 256 + threadIdx.x;    // B*N threads
  int b = idx >> 11, n = idx & 2047;
  const unsigned short* base = WhT + (((size_t)b * FOUT) << 11) + n;
  float s1 = 0.f, s2 = 0.f;
  for (int f = 0; f < FOUT; ++f) {
    float v = bf2f(base[(size_t)f << 11]);
    s1 += v * a[f];
    s2 += v * a[FOUT + f];
  }
  e1[idx] = s1;
  e2[idx] = s2;
}

// ---------------------------------------------------------------------------
// Kernel 3: fused mask + leakyrelu + softmax + (attn @ Wh).
// One workgroup (256 thr / 8 waves) per (batch, 16-row tile). adj read once.
// ---------------------------------------------------------------------------
__global__ void k_attn(const float* __restrict__ e1g, const float* __restrict__ e2g,
                       const int* __restrict__ adj,
                       const unsigned short* __restrict__ WhT,
                       float* __restrict__ out) {
  extern __shared__ char smem[];
  float*          eBuf = (float*)smem;                       // 16 x 2048 f32 = 128 KB
  float*          e2s  = (float*)(smem + 131072);            // 2048 f32     =   8 KB
  unsigned short* pBuf = (unsigned short*)(smem + 139264);   // 16 x 2088 bf16 ~ 65 KB
  float*          red  = (float*)(smem + 206080);            // 16 x 16 f32
  float*          mrow = red + 256;                          // 16
  float*          srow = mrow + 16;                          // 16
  float*          e1s  = srow + 16;                          // 16

  const int t  = threadIdx.x;
  const int b  = blockIdx.x >> 7;              // 128 row-tiles per batch
  const int i0 = (blockIdx.x & 127) << 4;

  // stage e2 row (8 KB) and the 16 e1 values
  for (int j = t; j < N_; j += 256) e2s[j] = e2g[(b << 11) + j];
  if (t < 16) e1s[t] = e1g[(b << 11) + i0 + t];
  __syncthreads();

  // ---- Phase 1: masked leaky scores into LDS + row max --------------------
  const int row = t >> 4, sub = t & 15;        // 16 threads per row
  const int jbase = sub * 128;                 // contiguous 128-col chunk
  const int* arow = adj + (((size_t)(b * N_ + i0 + row)) << 11) + jbase;
  float* erow = eBuf + (row << 11) + jbase;
  const float er1 = e1s[row];

  float lmax = NEGINF;
  for (int q = 0; q < 32; ++q) {               // 128 cols as int4/float4
    int4   av = ((const int4*)arow)[q];
    float4 ev = ((const float4*)(e2s + jbase))[q];
    float4 e;
    e.x = er1 + ev.x; e.x = e.x > 0.f ? e.x : ALPHA * e.x; e.x = av.x ? e.x : NEGINF;
    e.y = er1 + ev.y; e.y = e.y > 0.f ? e.y : ALPHA * e.y; e.y = av.y ? e.y : NEGINF;
    e.z = er1 + ev.z; e.z = e.z > 0.f ? e.z : ALPHA * e.z; e.z = av.z ? e.z : NEGINF;
    e.w = er1 + ev.w; e.w = e.w > 0.f ? e.w : ALPHA * e.w; e.w = av.w ? e.w : NEGINF;
    ((float4*)erow)[q] = e;
    lmax = fmaxf(lmax, fmaxf(fmaxf(e.x, e.y), fmaxf(e.z, e.w)));
  }
  red[(row << 4) + sub] = lmax;
  __syncthreads();
  if (t < 16) {
    float m = red[t << 4];
    for (int q = 1; q < 16; ++q) m = fmaxf(m, red[(t << 4) + q]);
    mrow[t] = m;
  }
  __syncthreads();

  // ---- Phase 1b: p = exp(e - m) -> bf16 LDS (padded rows), row sums -------
  {
    const float m = mrow[row];
    float ls = 0.f;
    uint4* pW = (uint4*)((unsigned*)pBuf + row * PSTRIDE_U32 + (jbase >> 1));
    for (int q4 = 0; q4 < 16; ++q4) {          // 128 cols as 16 x uint4
      Pack8 pk;
#pragma unroll
      for (int r = 0; r < 4; ++r) {
        float p0 = __expf(erow[8 * q4 + 2 * r]     - m);
        float p1 = __expf(erow[8 * q4 + 2 * r + 1] - m);
        ls += p0 + p1;
        pk.u[r] = pack_bf16(p0, p1);
      }
      pW[q4] = pk.q;
    }
    red[(row << 4) + sub] = ls;
  }
  __syncthreads();
  if (t < 16) {
    float s = 0.f;
    for (int q = 0; q < 16; ++q) s += red[(t << 4) + q];
    srow[t] = s;
  }
  __syncthreads();

  // ---- Phase 2: out[16 x 128] = (p/s) @ Wh via WMMA; wave w owns cols 16w --
  const int wave = t >> 5, lane = t & 31;
  const int hlf = lane >> 4, lm = lane & 15;
  const int fc = wave << 4;
  const unsigned* pRowU = (const unsigned*)pBuf + lm * PSTRIDE_U32;  // A row M=lm
  const unsigned short* bBase = WhT + (((size_t)(b * FOUT + fc + lm)) << 11);

  v8f acc = {};
  for (int kt = 0; kt < N_; kt += 32) {
    AFrag a;
    // two conflict-free 16B ds_load_b128 per fragment (see PSTRIDE comment)
    uint4 x = *(const uint4*)(pRowU + ((kt + 16 * hlf) >> 1));
    uint4 y = *(const uint4*)(pRowU + ((kt + 16 * hlf + 8) >> 1));
    a.u[0] = x.x; a.u[1] = x.y; a.u[2] = x.z; a.u[3] = x.w;
    a.u[4] = y.x; a.u[5] = y.y; a.u[6] = y.z; a.u[7] = y.w;
    const v16bf bf = *(const v16bf*)(bBase + kt + 16 * hlf);   // 32B contig per lane
    acc = __builtin_amdgcn_wmma_f32_16x16x32_bf16(
        false, a.v, false, bf, (short)0, acc, false, false);
  }

#pragma unroll
  for (int r = 0; r < 8; ++r) {
    const int M = r + 8 * hlf;
    out[(((size_t)(b * N_ + i0 + M)) << 7) + fc + lm] = acc[r] * (1.0f / srow[M]);
  }
}

// ---------------------------------------------------------------------------
extern "C" void kernel_launch(void* const* d_in, const int* in_sizes, int n_in,
                              void* d_out, int out_size, void* d_ws, size_t ws_size,
                              hipStream_t stream) {
  (void)in_sizes; (void)n_in; (void)out_size; (void)ws_size;
  const float* h   = (const float*)d_in[0];
  const int*   adj = (const int*)d_in[1];
  const float* W   = (const float*)d_in[2];
  const float* a   = (const float*)d_in[3];
  float* out = (float*)d_out;

  // workspace layout (~4.2 MB)
  char* ws = (char*)d_ws;
  unsigned short* WhT = (unsigned short*)ws;                       // 4 MiB
  unsigned short* WT  = (unsigned short*)(ws + (4u << 20));        // 64 KiB
  float* e1 = (float*)(ws + (4u << 20) + (64u << 10));             // 64 KiB
  float* e2 = (float*)(ws + (4u << 20) + (128u << 10));            // 64 KiB

  k_wt   <<<64,   256, 0, stream>>>(W, (unsigned*)WT);
  k_gemm1<<<128,  256, 0, stream>>>(h, WT, WhT);
  k_e    <<<64,   256, 0, stream>>>(WhT, a, e1, e2);
  // dynamic LDS: 131072 + 8192 + 66816 + 1024 + 3*64 = 207296 bytes (< 320 KB/WGP)
  k_attn <<<1024, 256, 207296, stream>>>(e1, e2, adj, WhT, out);
}
